// Conv2d_int8_STE_9354438770982
// MI455X (gfx1250) — 8-bit matrix operand order repaired and checked by reading, compiled, not run on hardware
//
#include <hip/hip_runtime.h>
#include <stdint.h>


typedef __attribute__((ext_vector_type(8))) int v8i;

#define NB    8
#define C_IN  32
#define C_OUT 32
#define HH    32
#define WW    32
#define KDIM  288
#define KPAD  320

__global__ void init_amax_kernel(uint32_t* amax) {
    if (threadIdx.x < 2) amax[threadIdx.x] = 0u;
}

__global__ void absmax_kernel(const float* __restrict__ p, int n,
                              uint32_t* __restrict__ slot) {
    __shared__ uint32_t sm[256];
    uint32_t local = 0u;
    for (int i = blockIdx.x * blockDim.x + threadIdx.x; i < n;
         i += gridDim.x * blockDim.x) {
        uint32_t b = __float_as_uint(fabsf(p[i]));
        local = local > b ? local : b;
    }
    sm[threadIdx.x] = local;
    __syncthreads();
    for (int s = blockDim.x >> 1; s > 0; s >>= 1) {
        if (threadIdx.x < (unsigned)s) {
            uint32_t o = sm[threadIdx.x + s];
            if (o > sm[threadIdx.x]) sm[threadIdx.x] = o;
        }
        __syncthreads();
    }
    if (threadIdx.x == 0) atomicMax(slot, sm[0]);
}

__global__ void quant_x_kernel(const float* __restrict__ x,
                               int8_t* __restrict__ qx,
                               const uint32_t* __restrict__ amax) {
    int idx = blockIdx.x * blockDim.x + threadIdx.x;
    if (idx >= NB * C_IN * HH * WW) return;
    float inv = 127.0f / __uint_as_float(amax[0]);
    float q = rintf(x[idx] * inv);
    q = fminf(fmaxf(q, -128.0f), 127.0f);
    int xw = idx & 31;
    int y  = (idx >> 5) & 31;
    int c  = (idx >> 10) & 31;
    int n  = idx >> 15;
    qx[(((n * HH + y) * WW + xw) * C_IN) + c] = (int8_t)(int)q;
}

__global__ void quant_w_kernel(const float* __restrict__ wgt,
                               int8_t* __restrict__ qw,
                               const uint32_t* __restrict__ amax) {
    int idx = blockIdx.x * blockDim.x + threadIdx.x;
    if (idx >= C_OUT * KPAD) return;
    int k = idx % KPAD;
    int o = idx / KPAD;
    int8_t q8 = 0;
    if (k < KDIM) {
        float inv = 127.0f / __uint_as_float(amax[1]);
        int t = k >> 5;
        int c = k & 31;
        int kh = t / 3, kw = t - kh * 3;
        float q = rintf(wgt[((o * C_IN + c) * 3 + kh) * 3 + kw] * inv);
        q = fminf(fmaxf(q, -128.0f), 127.0f);
        q8 = (int8_t)(int)q;
    }
    qw[idx] = q8;
}

__global__ void conv_wmma_iu8_kernel(const int8_t* __restrict__ qx,
                                     const int8_t* __restrict__ qw,
                                     const float* __restrict__ bias,
                                     const uint32_t* __restrict__ amax,
                                     float* __restrict__ out) {
    int wave = (blockIdx.x * blockDim.x + threadIdx.x) >> 5;
    int lane = threadIdx.x & 31;

    int w0   = (wave & 1) << 4;
    int h    = (wave >> 1) & 31;
    int n    = wave >> 6;
    int m    = lane & 15;
    int half = lane >> 4;
    int ocol = lane & 15;

    v8i acc0 = {};
    v8i acc1 = {};

    for (int kt = 0; kt < 5; ++kt) {
        v8i a;
#pragma unroll
        for (int v = 0; v < 8; ++v) {
            int k0 = ((v >> 1) << 4) + (half << 3) + ((v & 1) << 2);
            int t  = (kt << 1) + (k0 >> 5);
            int c0 = k0 & 31;
            uint32_t av = 0u;
            if (t < 9) {
                int dy = t / 3, dx = t - dy * 3;
                int yy = h + dy - 1;
                int xx = w0 + m + dx - 1;
                if ((unsigned)yy < 32u && (unsigned)xx < 32u) {
                    av = *(const uint32_t*)(qx +
                         (((n * HH + yy) * WW + xx) * C_IN + c0));
                }
            }
            a[v] = (int)av;
        }
        v8i b0, b1;
#pragma unroll
        for (int v = 0; v < 8; ++v) {
            int kb = ((v >> 1) << 4) + (half << 3) + ((v & 1) << 2);
            int k  = (kt << 6) + kb;
            b0[v] = *(const int*)(qw + ocol * KPAD + k);
            b1[v] = *(const int*)(qw + (ocol + 16) * KPAD + k);
        }
        acc0 = __builtin_amdgcn_wmma_i32_16x16x64_iu8(true, a, true, b0, acc0,
                                                      false, false);
        acc1 = __builtin_amdgcn_wmma_i32_16x16x64_iu8(true, a, true, b1, acc1,
                                                      false, false);
    }

    float sf = __uint_as_float(amax[0]) * (1.0f / 127.0f);
    float sw = __uint_as_float(amax[1]) * (1.0f / 127.0f);
    float scale = sf * sw;
    float b_lo = bias[ocol];
    float b_hi = bias[ocol + 16];

#pragma unroll
    for (int r = 0; r < 8; ++r) {
        int wp = w0 + r + (half << 3);
        out[((n * C_OUT + ocol)      * HH + h) * WW + wp] = scale * (float)acc0[r] + b_lo;
        out[((n * C_OUT + ocol + 16) * HH + h) * WW + wp] = scale * (float)acc1[r] + b_hi;
    }
}

extern "C" void kernel_launch(void* const* d_in, const int* in_sizes, int n_in,
                              void* d_out, int out_size, void* d_ws, size_t ws_size,
                              hipStream_t stream) {
    const float* x    = (const float*)d_in[0];
    const float* wgt  = (const float*)d_in[1];
    const float* bias = (const float*)d_in[2];
    float* out = (float*)d_out;

    uint32_t* amax = (uint32_t*)d_ws;
    int8_t*   qx   = (int8_t*)d_ws + 16;
    int8_t*   qw   = qx + NB * HH * WW * C_IN;

    const int nx = NB * C_IN * HH * WW;
    const int nw = C_OUT * C_IN * 9;

    init_amax_kernel<<<1, 32, 0, stream>>>(amax);
    absmax_kernel<<<256, 256, 0, stream>>>(x, nx, amax + 0);
    absmax_kernel<<<36, 256, 0, stream>>>(wgt, nw, amax + 1);
    quant_x_kernel<<<(nx + 255) / 256, 256, 0, stream>>>(x, qx, amax);
    quant_w_kernel<<<(C_OUT * KPAD + 255) / 256, 256, 0, stream>>>(wgt, qw, amax);
    conv_wmma_iu8_kernel<<<128, 128, 0, stream>>>(qx, qw, bias, amax, out);
}
